// NeuralGCDE_2937757630529
// MI455X (gfx1250) — matrix operand order repaired and checked by reading, not compiled, not run on hardware
//
#include <hip/hip_runtime.h>
#include <hip/hip_bf16.h>
#include <cstdint>

#define B_   16
#define N_   307
#define T_   12
#define C_   2
#define H_   64
#define HH_  64
#define ED_  8
#define HOR_ 12
#define M_   (N_*B_)   /* 4912 samples; s = n*B_ + b */

typedef __attribute__((ext_vector_type(16))) __bf16       v16bf;
typedef __attribute__((ext_vector_type(8)))  float        v8f;
typedef __attribute__((ext_vector_type(8)))  unsigned int v8u;

__device__ __forceinline__ unsigned int f2bf(float f) {
  unsigned int u = __builtin_bit_cast(unsigned int, f);
  return (u + 0x7FFFu + ((u >> 16) & 1u)) >> 16;   // round-to-nearest-even
}
__device__ __forceinline__ unsigned int packbf2(float lo, float hi) {
  return f2bf(lo) | (f2bf(hi) << 16);
}

// A-fragment (16x32 bf16) from row-major f32 LDS tile act[16][ld] (16B aligned).
// ISA layout: lanes 0-15 M=0-15 / 16-31 M=0-15; v0-3: K={0..7 | 8..15}, v4-7: K={16..23 | 24..31}
// k is always even -> one ds_load_b64 per packed dword.
__device__ __forceinline__ v16bf make_afrag(const float* act, int ld, int kbase, int lane) {
  int m = lane & 15;
  int khalf = (lane >> 4) * 8;
  v8u u;
#pragma unroll
  for (int v = 0; v < 8; ++v) {
    int k = kbase + ((v & 4) ? 16 : 0) + khalf + 2 * (v & 3);
    const float2 p = *(const float2*)(act + m * ld + k);
    u[v] = packbf2(p.x, p.y);
  }
  return __builtin_bit_cast(v16bf, u);
}

// B-fragment pre-packed in memory: [tile][kchunk][lane][v] dwords; one 32B load/lane.
__device__ __forceinline__ v16bf load_bfrag(const unsigned int* wp, int lane) {
  return __builtin_bit_cast(v16bf, *(const v8u*)(wp + lane * 8));
}

__device__ __forceinline__ v8f wmma_bf16(v16bf a, v16bf b, v8f c) {
  return __builtin_amdgcn_wmma_f32_16x16x32_bf16(false, a, false, b, (short)0, c, false, false);
}

// ---------------- setup kernels (run once per launch) ----------------

__global__ void k_softmaxA(const float* __restrict__ gE, float* __restrict__ A) {
  int n = blockIdx.x, tid = threadIdx.x;
  __shared__ float sc[N_];
  __shared__ float red[256];
  float en[ED_];
#pragma unroll
  for (int d = 0; d < ED_; ++d) en[d] = gE[n * ED_ + d];
  for (int m = tid; m < N_; m += 256) {
    float acc = 0.f;
#pragma unroll
    for (int d = 0; d < ED_; ++d) acc += en[d] * gE[m * ED_ + d];
    sc[m] = acc > 0.f ? acc : 0.f;
  }
  __syncthreads();
  float mx = -1e30f;
  for (int m = tid; m < N_; m += 256) mx = fmaxf(mx, sc[m]);
  red[tid] = mx; __syncthreads();
  for (int o = 128; o > 0; o >>= 1) { if (tid < o) red[tid] = fmaxf(red[tid], red[tid + o]); __syncthreads(); }
  mx = red[0]; __syncthreads();
  float sm = 0.f;
  for (int m = tid; m < N_; m += 256) sm += expf(sc[m] - mx);
  red[tid] = sm; __syncthreads();
  for (int o = 128; o > 0; o >>= 1) { if (tid < o) red[tid] += red[tid + o]; __syncthreads(); }
  float inv = 1.f / red[0];
  for (int m = tid; m < N_; m += 256) A[n * N_ + m] = expf(sc[m] - mx) * inv;
}

// Pack K x NC f32 row-major weight into WMMA-B bf16 fragments: [t][kc][lane][v]
__global__ void k_pack_b(unsigned int* __restrict__ dst, const float* __restrict__ src,
                         int K, int NC) {
  int kcN = K / 32;
  int total = (NC / 16) * kcN * 256;
  for (int idx = blockIdx.x * blockDim.x + threadIdx.x; idx < total;
       idx += gridDim.x * blockDim.x) {
    int v  = idx & 7;
    int L  = (idx >> 3) & 31;
    int blk = idx >> 8;            // blk = t*kcN + kc
    int kc = blk % kcN;
    int t  = blk / kcN;
    int k  = kc * 32 + ((v & 4) << 2) + ((L >> 4) << 3) + ((v & 3) << 1);
    int oc = t * 16 + (L & 15);
    dst[idx] = packbf2(src[k * NC + oc], src[(k + 1) * NC + oc]);
  }
}

// agc_W[n,k,i,o] = sum_d gE[n,d] * gWpool[d,k,i,o], packed directly to B-fragments.
__global__ void k_pack_agcw(unsigned int* __restrict__ dst, const float* __restrict__ gE,
                            const float* __restrict__ gWpool) {
  int total = N_ << 12;   // 307 * 2k * 4t * 2kc * 32L * 8v
  for (int idx = blockIdx.x * blockDim.x + threadIdx.x; idx < total;
       idx += gridDim.x * blockDim.x) {
    int v  = idx & 7;
    int L  = (idx >> 3) & 31;
    int kc = (idx >> 8) & 1;
    int t  = (idx >> 9) & 3;
    int k  = (idx >> 11) & 1;
    int n  = idx >> 12;
    int i  = kc * 32 + ((v & 4) << 2) + ((L >> 4) << 3) + ((v & 3) << 1);
    int o  = t * 16 + (L & 15);
    float w0 = 0.f, w1 = 0.f;
#pragma unroll
    for (int d = 0; d < ED_; ++d) {
      float e = gE[n * ED_ + d];
      w0 += e * gWpool[((d * 2 + k) * HH_ + i) * HH_ + o];
      w1 += e * gWpool[((d * 2 + k) * HH_ + i + 1) * HH_ + o];
    }
    dst[idx] = packbf2(w0, w1);
  }
}

// agc_b, h0, z0
__global__ void k_init(float* __restrict__ agcb, float* __restrict__ h0, float* __restrict__ z0,
                       const float* __restrict__ gE, const float* __restrict__ gbpool,
                       const float* __restrict__ coeff_a,
                       const float* __restrict__ hW, const float* __restrict__ hb,
                       const float* __restrict__ zW, const float* __restrict__ zb) {
  int tot0 = N_ * H_;
  int tot1 = M_ * H_;
  int total = tot0 + 2 * tot1;
  for (int idx = blockIdx.x * blockDim.x + threadIdx.x; idx < total;
       idx += gridDim.x * blockDim.x) {
    if (idx < tot0) {
      int n = idx / H_, o = idx % H_;
      float a = 0.f;
#pragma unroll
      for (int d = 0; d < ED_; ++d) a += gE[n * ED_ + d] * gbpool[d * H_ + o];
      agcb[idx] = a;
    } else {
      int e = idx - tot0;
      int which = e / tot1; e %= tot1;
      int s = e / H_, hh = e % H_;
      int n = s / B_, b = s % B_;
      float x0 = coeff_a[((b * N_ + n) * (T_ - 1)) * C_ + 0];
      float x1 = coeff_a[((b * N_ + n) * (T_ - 1)) * C_ + 1];
      if (which == 0) h0[e] = x0 * hW[hh] + x1 * hW[H_ + hh] + hb[hh];
      else            z0[e] = x0 * zW[hh] + x1 * zW[H_ + hh] + zb[hh];
    }
  }
}

// ---------------- per-RK4-stage kernels ----------------
// Stage kernel 1: h_in/z_in, func_f chain (3 WMMA GEMMs), dX, kh = f_v . dX,
// hacc update, and x = relu(z_in @ g_W_in + b) for the graph conv.
__global__ __launch_bounds__(256) void k_stage1(
    const float* __restrict__ h, float* __restrict__ hacc,
    float* __restrict__ khbuf, const float* __restrict__ kzbuf,
    const float* __restrict__ z, float* __restrict__ xout,
    const unsigned int* __restrict__ fwin_p,  const float* __restrict__ f_b_in,
    const unsigned int* __restrict__ fwmid_p, const float* __restrict__ f_b_mid,
    const unsigned int* __restrict__ fwout_p, const float* __restrict__ f_b_out,
    const unsigned int* __restrict__ gwin_p,  const float* __restrict__ g_b_in,
    const float* __restrict__ cb, const float* __restrict__ cc, const float* __restrict__ cd,
    int idx, float frac, float a_s, float w_s, int stage)
{
  __shared__ __align__(16) float hin[16 * 64], zin[16 * 64], actA[16 * 64], actB[16 * 64];
  __shared__ __align__(16) float fv[16 * 128];
  __shared__ float dXs[16 * 2];
  int n = blockIdx.x, tid = threadIdx.x;
  int wave = tid >> 5, lane = tid & 31;

  if (tid < 32) {                               // spline derivative (idx/frac are host consts)
    int b = tid >> 1, c = tid & 1;
    int off = ((b * N_ + n) * (T_ - 1) + idx) * C_ + c;
    dXs[b * 2 + c] = cb[off] + (cc[off] + cd[off] * frac) * frac;
  }
  for (int e = tid; e < 1024; e += 256) {       // h_in = h + a*k_prev, z_in = z + a*k_prev
    int s = n * B_ + (e >> 6);
    int col = e & 63;
    float hv = h[s * 64 + col], zv = z[s * 64 + col];
    if (stage > 0) { hv += a_s * khbuf[s * 64 + col]; zv += a_s * kzbuf[s * 64 + col]; }
    hin[e] = hv; zin[e] = zv;
  }
  __syncthreads();

  // Wave-uniform split (keeps EXEC all-ones through WMMA, no per-element saveexec):
  if (wave < 4) {   // waves 0-3: t1 = relu(hin @ f_W_in + b) -> actA (LDS)
    int t = wave;
    v8f c = {};
#pragma unroll
    for (int kc = 0; kc < 2; ++kc)
      c = wmma_bf16(make_afrag(hin, 64, kc * 32, lane),
                    load_bfrag(fwin_p + (t * 2 + kc) * 256, lane), c);
    int col = lane & 15, mb = (lane >> 4) * 8, oc = t * 16 + col;
    float bia = f_b_in[oc];
#pragma unroll
    for (int v = 0; v < 8; ++v) actA[(mb + v) * 64 + oc] = fmaxf(c[v] + bia, 0.f);
  } else {          // waves 4-7: x = relu(zin @ g_W_in + b) -> global
    int t = wave - 4;
    v8f c = {};
#pragma unroll
    for (int kc = 0; kc < 2; ++kc)
      c = wmma_bf16(make_afrag(zin, 64, kc * 32, lane),
                    load_bfrag(gwin_p + (t * 2 + kc) * 256, lane), c);
    int col = lane & 15, mb = (lane >> 4) * 8, oc = t * 16 + col;
    float bia = g_b_in[oc];
#pragma unroll
    for (int v = 0; v < 8; ++v) xout[(n * B_ + mb + v) * 64 + oc] = fmaxf(c[v] + bia, 0.f);
  }
  __syncthreads();

  if (wave < 4) {  // t2 = relu(t1 @ f_W_mid + b) -> actB
    int t = wave;
    v8f c = {};
#pragma unroll
    for (int kc = 0; kc < 2; ++kc)
      c = wmma_bf16(make_afrag(actA, 64, kc * 32, lane),
                    load_bfrag(fwmid_p + (t * 2 + kc) * 256, lane), c);
    int col = lane & 15, mb = (lane >> 4) * 8, oc = t * 16 + col;
    float bia = f_b_mid[oc];
#pragma unroll
    for (int v = 0; v < 8; ++v) actB[(mb + v) * 64 + oc] = fmaxf(c[v] + bia, 0.f);
  }
  __syncthreads();

  { // f_v = tanh(t2 @ f_W_out + b) : 128 cols -> 8 tiles, one per wave
    int t = wave;
    v8f c = {};
#pragma unroll
    for (int kc = 0; kc < 2; ++kc)
      c = wmma_bf16(make_afrag(actB, 64, kc * 32, lane),
                    load_bfrag(fwout_p + (t * 2 + kc) * 256, lane), c);
    int col = lane & 15, mb = (lane >> 4) * 8, oc = t * 16 + col;
    float bia = f_b_out[oc];
#pragma unroll
    for (int v = 0; v < 8; ++v) fv[(mb + v) * 128 + oc] = tanhf(c[v] + bia);
  }
  __syncthreads();

  for (int e = tid; e < 1024; e += 256) {       // kh = f_v . dX ; RK4 accumulate
    int m = e >> 6, hh = e & 63;
    int s = n * B_ + m;
    float kv = fv[m * 128 + 2 * hh] * dXs[m * 2 + 0] + fv[m * 128 + 2 * hh + 1] * dXs[m * 2 + 1];
    khbuf[s * 64 + hh] = kv;
    float base = (stage == 0) ? h[s * 64 + hh] : hacc[s * 64 + hh];
    hacc[s * 64 + hh] = base + w_s * kv;
  }
}

// Stage kernel 2: graph conv, per-node AGC mixing (WMMA), then the dominant GEMM
// (y @ g_W_out, 64->4096) fused with tanh and the contraction against kh -> kz.
__global__ __launch_bounds__(256) void k_stage2(
    const float* __restrict__ z, float* __restrict__ zacc,
    const float* __restrict__ khbuf, float* __restrict__ kzbuf,
    const float* __restrict__ x, const float* __restrict__ Amat,
    const unsigned int* __restrict__ agcw_p, const float* __restrict__ agcb,
    const unsigned int* __restrict__ gwout_p, const float* __restrict__ g_b_out,
    float w_s, int stage)
{
  __shared__ __align__(16) float xn[1024], xa[1024], y0[1024], y1[1024], yb[1024], khl[1024];
  __shared__ float Arow[N_];
  int n = blockIdx.x, tid = threadIdx.x;
  int wave = tid >> 5, lane = tid & 31;

  for (int m = tid; m < N_; m += 256) Arow[m] = Amat[n * N_ + m];
  for (int e = tid; e < 1024; e += 256) {
    int s = n * B_ + (e >> 6);
    xn[e]  = x[s * 64 + (e & 63)];
    khl[e] = khbuf[s * 64 + (e & 63)];
  }
  __syncthreads();

  // xa[b,i] = sum_m A[n,m] * x[b,m,i]  (batched matvec; x stays L2-resident)
  for (int e = tid; e < 1024; e += 256) {
    int b = e >> 6, i = e & 63;
    float acc = 0.f;
    for (int mm = 0; mm < N_; ++mm) acc += Arow[mm] * x[(mm * B_ + b) * 64 + i];
    xa[e] = acc;
  }
  __syncthreads();

  { // y partials: waves 0-3 identity-term, waves 4-7 A-term, per-node agc weights
    int k = wave >> 2;
    int t = wave & 3;
    const float* src = k ? xa : xn;
    float* dst = k ? y1 : y0;
    v8f c = {};
#pragma unroll
    for (int kc = 0; kc < 2; ++kc)
      c = wmma_bf16(make_afrag(src, 64, kc * 32, lane),
                    load_bfrag(agcw_p + (((n * 2 + k) * 4 + t) * 2 + kc) * 256, lane), c);
    int col = lane & 15, mb = (lane >> 4) * 8, oc = t * 16 + col;
#pragma unroll
    for (int v = 0; v < 8; ++v) dst[(mb + v) * 64 + oc] = c[v];
  }
  __syncthreads();
  for (int e = tid; e < 1024; e += 256)
    yb[e] = fmaxf(y0[e] + y1[e] + agcb[n * 64 + (e & 63)], 0.f);
  __syncthreads();

  // fused: kz[s,i] = sum_j tanh((yb @ g_W_out + b)[s, i*64+j]) * kh[s,j]
  v16bf ay0 = make_afrag(yb, 64, 0,  lane);
  v16bf ay1 = make_afrag(yb, 64, 32, lane);
  int col = lane & 15, mb = (lane >> 4) * 8;
#pragma unroll 1
  for (int ii = 0; ii < 8; ++ii) {
    int i = wave * 8 + ii;
    if (ii + 1 < 8) {   // prefetch next i-block's B-fragments (global_prefetch_b8)
      const unsigned int* np = gwout_p + (size_t)(i + 1) * 8 * 256 + lane * 8;
      __builtin_prefetch(np, 0, 1);
      __builtin_prefetch(np + 1024, 0, 1);
    }
    float part[8] = {0.f, 0.f, 0.f, 0.f, 0.f, 0.f, 0.f, 0.f};
#pragma unroll 1
    for (int jt = 0; jt < 4; ++jt) {
      int t = i * 4 + jt;
      v8f c = {};
      c = wmma_bf16(ay0, load_bfrag(gwout_p + (t * 2 + 0) * 256, lane), c);
      c = wmma_bf16(ay1, load_bfrag(gwout_p + (t * 2 + 1) * 256, lane), c);
      float bia = g_b_out[t * 16 + col];
      int j = jt * 16 + col;
#pragma unroll
      for (int v = 0; v < 8; ++v)
        part[v] += tanhf(c[v] + bia) * khl[(mb + v) * 64 + j];
    }
#pragma unroll
    for (int off = 1; off < 16; off <<= 1)
#pragma unroll
      for (int v = 0; v < 8; ++v)
        part[v] += __shfl_xor(part[v], off, 32);
    if (col == 0) {
#pragma unroll
      for (int v = 0; v < 8; ++v) {
        int s = n * B_ + mb + v;
        float kv = part[v];
        kzbuf[s * 64 + i] = kv;
        float base = (stage == 0) ? z[s * 64 + i] : zacc[s * 64 + i];
        zacc[s * 64 + i] = base + w_s * kv;
      }
    }
  }
}

__global__ void k_out(const float* __restrict__ zT, const float* __restrict__ convW,
                      const float* __restrict__ convb, float* __restrict__ out) {
  int total = B_ * HOR_ * N_;
  for (int idx = blockIdx.x * blockDim.x + threadIdx.x; idx < total;
       idx += gridDim.x * blockDim.x) {
    int n   = idx % N_;
    int hor = (idx / N_) % HOR_;
    int b   = idx / (N_ * HOR_);
    float acc = convb[hor];
    const float* zr = zT + (n * B_ + b) * 64;
    const float* wr = convW + hor * 64;
#pragma unroll
    for (int hh = 0; hh < 64; ++hh) acc += zr[hh] * wr[hh];
    out[idx] = acc;
  }
}

// ---------------- host ----------------
extern "C" void kernel_launch(void* const* d_in, const int* in_sizes, int n_in,
                              void* d_out, int out_size, void* d_ws, size_t ws_size,
                              hipStream_t stream) {
  (void)in_sizes; (void)n_in; (void)out_size; (void)ws_size;
  const float* coeff_a = (const float*)d_in[0];
  const float* coeff_b = (const float*)d_in[1];
  const float* coeff_c = (const float*)d_in[2];
  const float* coeff_d = (const float*)d_in[3];
  /* times = d_in[4] = arange(12), folded into host constants */
  const float* h_W     = (const float*)d_in[5];
  const float* h_b     = (const float*)d_in[6];
  const float* z_W     = (const float*)d_in[7];
  const float* z_b     = (const float*)d_in[8];
  const float* f_W_in  = (const float*)d_in[9];
  const float* f_b_in  = (const float*)d_in[10];
  const float* f_W_mid = (const float*)d_in[11];
  const float* f_b_mid = (const float*)d_in[12];
  const float* f_W_out = (const float*)d_in[13];
  const float* f_b_out = (const float*)d_in[14];
  const float* g_W_in  = (const float*)d_in[15];
  const float* g_b_in  = (const float*)d_in[16];
  const float* g_E     = (const float*)d_in[17];
  const float* g_Wpool = (const float*)d_in[18];
  const float* g_bpool = (const float*)d_in[19];
  const float* g_W_out = (const float*)d_in[20];
  const float* g_b_out = (const float*)d_in[21];
  const float* conv_W  = (const float*)d_in[22];
  const float* conv_b  = (const float*)d_in[23];

  char* ws = (char*)d_ws;
  size_t off = 0;
  // 256B-align every allocation so 32B v8u fragment loads are always aligned.
  auto alloc_f = [&](size_t nfl) -> float* {
    float* p = (float*)(ws + off); off += (nfl * 4 + 255) & ~(size_t)255; return p; };
  auto alloc_u = [&](size_t ndw) -> unsigned int* {
    unsigned int* p = (unsigned int*)(ws + off); off += (ndw * 4 + 255) & ~(size_t)255; return p; };
  const size_t M64 = (size_t)M_ * 64;
  float* hA = alloc_f(M64);  float* hB = alloc_f(M64);
  float* zA = alloc_f(M64);  float* zB = alloc_f(M64);
  float* kh = alloc_f(M64);  float* kz = alloc_f(M64);
  float* xb = alloc_f(M64);
  float* Amat = alloc_f((size_t)N_ * N_);
  float* agcb = alloc_f((size_t)N_ * 64);
  unsigned int* fwin_p  = alloc_u(2048);
  unsigned int* fwmid_p = alloc_u(2048);
  unsigned int* fwout_p = alloc_u(4096);
  unsigned int* gwin_p  = alloc_u(2048);
  unsigned int* gwout_p = alloc_u(131072);
  unsigned int* agcw_p  = alloc_u((size_t)N_ << 12);

  k_softmaxA<<<N_, 256, 0, stream>>>(g_E, Amat);
  k_pack_b<<<8, 256, 0, stream>>>(fwin_p,  f_W_in,  64, 64);
  k_pack_b<<<8, 256, 0, stream>>>(fwmid_p, f_W_mid, 64, 64);
  k_pack_b<<<16, 256, 0, stream>>>(fwout_p, f_W_out, 64, 128);
  k_pack_b<<<8, 256, 0, stream>>>(gwin_p,  g_W_in,  64, 64);
  k_pack_b<<<512, 256, 0, stream>>>(gwout_p, g_W_out, 64, 4096);
  k_pack_agcw<<<1024, 256, 0, stream>>>(agcw_p, g_E, g_Wpool);
  k_init<<<1024, 256, 0, stream>>>(agcb, hA, zA, g_E, g_bpool, coeff_a, h_W, h_b, z_W, z_b);

  float* hc = hA; float* ha = hB; float* zc = zA; float* za = zB;
  const float aS[4] = {0.f, 0.5f, 0.5f, 1.f};
  const float wS[4] = {1.f / 6.f, 2.f / 6.f, 2.f / 6.f, 1.f / 6.f};
  for (int step = 0; step < T_ - 1; ++step) {
    for (int s = 0; s < 4; ++s) {
      int idx; float frac;
      if (s == 0)      { idx = step; frac = 0.f; }
      else if (s == 3) { idx = (step + 1 <= T_ - 2) ? step + 1 : T_ - 2; frac = (float)(step + 1 - idx); }
      else             { idx = step; frac = 0.5f; }
      k_stage1<<<N_, 256, 0, stream>>>(hc, ha, kh, kz, zc, xb,
          fwin_p, f_b_in, fwmid_p, f_b_mid, fwout_p, f_b_out, gwin_p, g_b_in,
          coeff_b, coeff_c, coeff_d, idx, frac, aS[s], wS[s], s);
      k_stage2<<<N_, 256, 0, stream>>>(zc, za, kh, kz, xb, Amat, agcw_p, agcb,
          gwout_p, g_b_out, wS[s], s);
    }
    float* t = hc; hc = ha; ha = t;
    t = zc; zc = za; za = t;
  }
  k_out<<<232, 256, 0, stream>>>(zc, conv_W, conv_b, (float*)d_out);
}
